// Model_37495064494703
// MI455X (gfx1250) — matrix operand order repaired and checked by reading, compiled, not run on hardware
//
#include <hip/hip_runtime.h>

#define DIM   512
#define KCB   1024
#define KS    320
#define TT    500
#define BATCH 16
#define NROW  (BATCH * TT)   // 8000 frames

typedef __attribute__((ext_vector_type(16))) __bf16 v16bf;
typedef __attribute__((ext_vector_type(8)))  float  v8f;

union BF16x16 {
    unsigned short s[16];
    uint4          q[2];
    v16bf          v;
};

__device__ __forceinline__ unsigned short f2bf(float f) {
    union { float f; unsigned u; } x; x.f = f;
    unsigned r = x.u + 0x7FFFu + ((x.u >> 16) & 1u);  // round-to-nearest-even
    return (unsigned short)(r >> 16);
}

// Monotonic float -> u32 key (total order matching float <)
__device__ __forceinline__ unsigned fkey(float f) {
    union { float f; unsigned u; } x; x.f = f;
    return (x.u & 0x80000000u) ? ~x.u : (x.u | 0x80000000u);
}

// ---------------- one-shot prep: f32 -> bf16, packed (4 elems/thread) ----------------
__global__ __launch_bounds__(256) void k_cvt(const float* __restrict__ src,
                                             unsigned short* __restrict__ dst, int n4) {
    int i = blockIdx.x * blockDim.x + threadIdx.x;
    if (i >= n4) return;
    float4 v = ((const float4*)src)[i];
    union { unsigned short s[4]; uint2 u; } o;
    o.s[0] = f2bf(v.x); o.s[1] = f2bf(v.y); o.s[2] = f2bf(v.z); o.s[3] = f2bf(v.w);
    ((uint2*)dst)[i] = o.u;
}

// ---------------- one-shot prep: transpose + f32 -> bf16 ----------------
// src[r*cols + c]  ->  dst[c*rows + r]   (coalesced writes)
__global__ __launch_bounds__(256) void k_tcvt(const float* __restrict__ src,
                                              unsigned short* __restrict__ dst,
                                              int rows, int cols) {
    int i = blockIdx.x * blockDim.x + threadIdx.x;
    if (i >= rows * cols) return;
    int c = i / rows, r = i - c * rows;
    dst[(size_t)c * rows + r] = f2bf(src[(size_t)r * cols + c]);
}

// ---------------- codebook squared norms ----------------
__global__ __launch_bounds__(256) void k_cbnorm(const float* __restrict__ cb,
                                                float* __restrict__ cnorm) {
    int k = blockIdx.x * blockDim.x + threadIdx.x;
    if (k >= KCB) return;
    const float4* p = (const float4*)(cb + (size_t)k * DIM);
    float s = 0.f;
    #pragma unroll 4
    for (int i = 0; i < DIM / 4; ++i) {
        float4 v = p[i];
        s += v.x * v.x + v.y * v.y + v.z * v.z + v.w * v.w;
    }
    cnorm[k] = s;
}

// ---------------- encoder: frames[8000x320] x enc_w^T[320x512] -> emb ----------------
__global__ __launch_bounds__(128) void k_encoder(const unsigned short* __restrict__ wavbf,
                                                 const unsigned short* __restrict__ ewt,
                                                 const float* __restrict__ enc_b,
                                                 float* __restrict__ emb,
                                                 unsigned short* __restrict__ embbf) {
    const int lane = threadIdx.x & 31;
    const int wave = threadIdx.x >> 5;
    const int mt   = blockIdx.x;               // 0..499  (16-row tile)
    const int nt   = blockIdx.y * 4 + wave;    // 0..31   (16-col tile)
    const int hi   = (lane >> 4) & 1;
    const int lpos = lane & 15;
    const int mrow = mt * 16 + lpos;

    v8f acc = {};
    for (int kc = 0; kc < KS / 32; ++kc) {
        // A fragment: lane holds row M=lpos; K per ISA 16-bit 16x32 layout
        BF16x16 a;
        const uint4* ap = (const uint4*)(wavbf + (size_t)mrow * KS + kc * 32 + hi * 8);
        __builtin_prefetch(ap + 4, 0, 1);      // next K-chunk
        a.q[0] = ap[0];                        // K = base .. base+7
        a.q[1] = ap[2];                        // K = base+16 .. base+23
        // B fragment: lane holds column N=lpos; K per ISA 16-bit 32x16 layout (strided gather from ewt)
        BF16x16 bm;
        const unsigned short* bp = ewt + (size_t)(kc * 32 + hi * 8) * DIM + nt * 16 + lpos;
        #pragma unroll
        for (int s = 0; s < 8; ++s) { bm.s[s] = bp[(size_t)s * DIM]; bm.s[s + 8] = bp[(size_t)(s + 16) * DIM]; }
        acc = __builtin_amdgcn_wmma_f32_16x16x32_bf16(false, a.v, false, bm.v,
                                                      (short)0, acc, false, false);
    }
    const int   ncol = nt * 16 + lpos;
    const float bias = enc_b[ncol];
    #pragma unroll
    for (int r = 0; r < 8; ++r) {
        const int   row = mt * 16 + (hi ? r + 8 : r);
        const float v   = acc[r] + bias;
        emb[(size_t)row * DIM + ncol]   = v;
        embbf[(size_t)row * DIM + ncol] = f2bf(v);
    }
}

// ---------------- pooled cond + FiLM: gamma/beta per (b,d) ----------------
__global__ __launch_bounds__(512) void k_cond(const float* __restrict__ emb,
                                              const float* __restrict__ lin_w,
                                              const float* __restrict__ lin_b,
                                              const float* __restrict__ film_w,
                                              const float* __restrict__ film_b,
                                              float* __restrict__ gws,
                                              float* __restrict__ bws) {
    __shared__ float ta[DIM];
    __shared__ float cond[DIM];
    const int b = blockIdx.x;
    const int d = threadIdx.x;

    float s = 0.f;
    const float* ep = emb + (size_t)(b * TT) * DIM + d;
    for (int t = 0; t < TT; ++t) s += ep[(size_t)t * DIM];
    ta[d] = s * (1.0f / TT);
    __syncthreads();

    float c = lin_b[d];
    const float* lw = lin_w + (size_t)d * DIM;
    #pragma unroll 4
    for (int j = 0; j < DIM; ++j) c += lw[j] * ta[j];
    cond[d] = c;
    __syncthreads();

    float g  = film_b[d];
    float be = film_b[DIM + d];
    const float* fg = film_w + (size_t)d * DIM;
    const float* fb = film_w + (size_t)(DIM + d) * DIM;
    #pragma unroll 4
    for (int j = 0; j < DIM; ++j) {
        const float cj = cond[j];
        g  += fg[j] * cj;
        be += fb[j] * cj;
    }
    gws[b * DIM + d] = g;
    bws[b * DIM + d] = be;
}

// ---------------- fused VQ argmin + gather + FiLM + decoder GEMM ----------------
__global__ __launch_bounds__(256) void k_vq_dec(const unsigned short* __restrict__ embbf,
                                                const float* __restrict__ cb,
                                                const unsigned short* __restrict__ cbt,
                                                const float* __restrict__ cnorm,
                                                const float* __restrict__ gws,
                                                const float* __restrict__ bws,
                                                const unsigned short* __restrict__ dwbf,
                                                const float* __restrict__ dec_b,
                                                float* __restrict__ out) {
    __shared__ unsigned short      abf[16 * DIM];   // 16 KB bf16 tile (reused)
    __shared__ unsigned long long  best[16];
    __shared__ unsigned            codes[16];

    const int tid  = threadIdx.x;
    const int lane = tid & 31;
    const int wave = tid >> 5;       // 0..7
    const int tile = blockIdx.x;     // 0..499
    const int row0 = tile * 16;
    const int hi   = (lane >> 4) & 1;
    const int lpos = lane & 15;

    // stage emb tile (already bf16) into LDS via the async global->LDS engine:
    // no VGPR bounce; tracked on ASYNCcnt, drained before the barrier.
    {
        // low 32 bits of a generic pointer into LDS == wave-relative LDS byte address
        const unsigned       lds0 = (unsigned)(size_t)(void*)abf;
        const unsigned char* g0   = (const unsigned char*)(embbf + (size_t)row0 * DIM);
        // 16KB tile, 16B per lane per issue: 4 issues of 256 threads
        #pragma unroll
        for (int i = 0; i < 4; ++i) {
            const unsigned       ldsb = lds0 + (i * 256 + tid) * 16;
            const unsigned char* gsrc = g0 + (i * 256 + tid) * 16;
            asm volatile("global_load_async_to_lds_b128 %0, %1, off"
                         :: "v"(ldsb), "v"(gsrc) : "memory");
        }
        asm volatile("s_wait_asynccnt 0" ::: "memory");
    }
    if (tid < 16) best[tid] = ~0ULL;
    __syncthreads();

    // ---- VQ: scores vs all 1024 codes; per-lane running min ----
    unsigned long long bk[8];
    #pragma unroll
    for (int r = 0; r < 8; ++r) bk[r] = ~0ULL;

    for (int j = 0; j < 8; ++j) {
        const int kt = wave + j * 8;   // code-tile 0..63
        v8f acc = {};
        for (int kc = 0; kc < DIM / 32; ++kc) {
            BF16x16 a;
            const uint4* lp = (const uint4*)(abf + lpos * DIM + kc * 32 + hi * 8);
            a.q[0] = lp[0];
            a.q[1] = lp[2];
            BF16x16 bm;
            const unsigned short* bp = cbt + (size_t)(kc * 32 + hi * 8) * KCB + kt * 16 + lpos;
            #pragma unroll
            for (int s = 0; s < 8; ++s) { bm.s[s] = bp[(size_t)s * KCB]; bm.s[s + 8] = bp[(size_t)(s + 16) * KCB]; }
            acc = __builtin_amdgcn_wmma_f32_16x16x32_bf16(false, a.v, false, bm.v,
                                                          (short)0, acc, false, false);
        }
        const int   kcode = kt * 16 + lpos;
        const float cn    = cnorm[kcode];
        #pragma unroll
        for (int r = 0; r < 8; ++r) {
            const float dist = cn - 2.0f * acc[r];
            const unsigned long long key =
                ((unsigned long long)fkey(dist) << 32) | (unsigned)kcode;
            bk[r] = (key < bk[r]) ? key : bk[r];
        }
    }
    #pragma unroll
    for (int r = 0; r < 8; ++r)
        atomicMin(&best[hi ? r + 8 : r], bk[r]);   // ds_min_u64, order-independent
    __syncthreads();
    if (tid < 16) codes[tid] = (unsigned)(best[tid] & 0xFFFFFFFFu);
    __syncthreads();

    // ---- rebuild LDS tile as modulated = gamma*codebook[code] + beta (bf16) ----
    for (int e = tid; e < 16 * DIM; e += 256) {
        const int m = e >> 9;
        const int d = e & (DIM - 1);
        const int b = (row0 + m) / TT;
        const float v = gws[b * DIM + d] * cb[(size_t)codes[m] * DIM + d] + bws[b * DIM + d];
        abf[e] = f2bf(v);
    }
    __syncthreads();

    // ---- decoder GEMM: [16x512] x dec_w[512x320] -> out frames ----
    const float db = dec_b[0];
    for (int nt = wave; nt < KS / 16; nt += 8) {
        v8f acc = {};
        for (int kc = 0; kc < DIM / 32; ++kc) {
            BF16x16 a;
            const uint4* lp = (const uint4*)(abf + lpos * DIM + kc * 32 + hi * 8);
            a.q[0] = lp[0];
            a.q[1] = lp[2];
            BF16x16 bm;
            const unsigned short* bp = dwbf + (size_t)(kc * 32 + hi * 8) * KS + nt * 16 + lpos;
            #pragma unroll
            for (int s = 0; s < 8; ++s) { bm.s[s] = bp[(size_t)s * KS]; bm.s[s + 8] = bp[(size_t)(s + 16) * KS]; }
            acc = __builtin_amdgcn_wmma_f32_16x16x32_bf16(false, a.v, false, bm.v,
                                                          (short)0, acc, false, false);
        }
        const int col = nt * 16 + lpos;
        #pragma unroll
        for (int r = 0; r < 8; ++r) {
            const int mrow = hi ? r + 8 : r;
            out[(size_t)(row0 + mrow) * KS + col] = acc[r] + db;
        }
    }
}

extern "C" void kernel_launch(void* const* d_in, const int* in_sizes, int n_in,
                              void* d_out, int out_size, void* d_ws, size_t ws_size,
                              hipStream_t stream) {
    (void)in_sizes; (void)n_in; (void)out_size; (void)ws_size;
    const float* wav    = (const float*)d_in[0];
    const float* enc_w  = (const float*)d_in[1];
    const float* enc_b  = (const float*)d_in[2];
    const float* cb     = (const float*)d_in[3];
    const float* lin_w  = (const float*)d_in[4];
    const float* lin_b  = (const float*)d_in[5];
    const float* film_w = (const float*)d_in[6];
    const float* film_b = (const float*)d_in[7];
    const float* dec_w  = (const float*)d_in[8];
    const float* dec_b  = (const float*)d_in[9];
    float* out = (float*)d_out;

    // workspace layout (all 16B aligned given 256B-aligned base)
    float*          emb   = (float*)d_ws;                       // 8000*512 f32
    unsigned short* embbf = (unsigned short*)(emb + (size_t)NROW * DIM); // 8000*512 bf16
    unsigned short* wavbf = embbf + (size_t)NROW * DIM;         // 8000*320 bf16
    unsigned short* ewt   = wavbf + (size_t)NROW * KS;          // [320][512] bf16
    unsigned short* cbt   = ewt + (size_t)KS * DIM;             // [512][1024] bf16
    unsigned short* dwbf  = cbt + (size_t)DIM * KCB;            // [512][320] bf16
    float*          cnorm = (float*)(dwbf + (size_t)DIM * KS);  // 1024 f32
    float*          gws   = cnorm + KCB;                        // 16*512
    float*          bws   = gws + BATCH * DIM;                  // 16*512

    // one-shot conversions / transposes
    k_cvt   <<<dim3((NROW * KS / 4 + 255) / 256), dim3(256), 0, stream>>>(wav, wavbf, NROW * KS / 4);
    k_cvt   <<<dim3((DIM * KS / 4 + 255) / 256),  dim3(256), 0, stream>>>(dec_w, dwbf, DIM * KS / 4);
    k_tcvt  <<<dim3((DIM * KS + 255) / 256),      dim3(256), 0, stream>>>(enc_w, ewt, DIM, KS);
    k_tcvt  <<<dim3((KCB * DIM + 255) / 256),     dim3(256), 0, stream>>>(cb, cbt, KCB, DIM);
    k_cbnorm<<<dim3(4),                           dim3(256), 0, stream>>>(cb, cnorm);

    // pipeline
    k_encoder<<<dim3(500, 8), dim3(128), 0, stream>>>(wavbf, ewt, enc_b, emb, embbf);
    k_cond   <<<dim3(BATCH),  dim3(512), 0, stream>>>(emb, lin_w, lin_b,
                                                      film_w, film_b, gws, bws);
    k_vq_dec <<<dim3(500),    dim3(256), 0, stream>>>(embbf, cb, cbt, cnorm, gws, bws,
                                                      dwbf, dec_b, out);
}
